// ST_21560735826083
// MI455X (gfx1250) — compile-verified, matrix operand order checked by reading, not run on hardware
//
#include <hip/hip_runtime.h>
#include <math.h>

#define NWAY 5
#define BB 2
#define QQ 15
#define UU 20
#define CC 64
#define HWP 441
#define MS 441
#define MU 8820          // UU*HWP
#define NCOL1 2205       // NWAY*MS
#define CH1 69           // ceil(MU/128)
#define LCAP 2205        // max possible L (<= one mutual per support col)
#define LTMAX 2646       // MS + LCAP
#define CH2 4            // ceil(HWP/128)
#define NEG_INF (-3.0e38f)

typedef __bf16 bf16_t;
typedef bf16_t v16bf __attribute__((ext_vector_type(16)));
typedef float v8f __attribute__((ext_vector_type(8)));

static __device__ inline int imin(int a, int b) { return a < b ? a : b; }

// Build one WMMA bf16 operand (A or B) from strided fp32 memory. Branch-free:
// callers clamp the row/col index so every address is in-bounds; out-of-range
// lanes carry (finite) duplicated data that downstream guards never observe.
// Per-lane K pattern for 16-bit 16x16x32 operands: K = (e<8 ? e : e+8) + 8*hi.
static __device__ inline v16bf load_bf16_tile(const float* base, int kbase, int stride, int hi) {
  v16bf v;
#pragma unroll
  for (int e = 0; e < 16; ++e) {
    int K = kbase + ((e < 8) ? e : e + 8) + (hi ? 8 : 0);
    v[e] = (bf16_t)base[(size_t)K * stride];
  }
  return v;
}

static __device__ inline v8f wmma_bf16(v16bf a, v16bf b, v8f c) {
  return __builtin_amdgcn_wmma_f32_16x16x32_bf16(false, a, false, b, (short)0, c, false, false);
}

// ---------------- Stage 0: channel-L2 normalization ----------------
__global__ void k_norm_sup(const float* __restrict__ sxf, float* __restrict__ sup_n) {
  int idx = blockIdx.x * blockDim.x + threadIdx.x;
  if (idx >= BB * NWAY * MS) return;
  int s = idx % MS, bn = idx / MS;
  const float* p = sxf + ((size_t)bn * CC) * MS + s;
  float ss = 0.f;
  for (int c = 0; c < CC; ++c) { float x = p[c * MS]; ss += x * x; }
  float inv = 1.0f / fmaxf(sqrtf(ss), 1e-12f);
  float* o = sup_n + ((size_t)bn * CC) * MS + s;
  for (int c = 0; c < CC; ++c) o[c * MS] = p[c * MS] * inv;
}

__global__ void k_norm_unl(const float* __restrict__ uxf, float* __restrict__ unl_n) {
  int idx = blockIdx.x * blockDim.x + threadIdx.x;
  if (idx >= BB * MU) return;
  int b = idx / MU, m = idx % MU;
  int u = m / HWP, p = m % HWP;
  const float* src = uxf + ((size_t)(b * UU + u) * CC) * HWP + p;
  float ss = 0.f;
  for (int c = 0; c < CC; ++c) { float x = src[c * HWP]; ss += x * x; }
  float inv = 1.0f / fmaxf(sqrtf(ss), 1e-12f);
  float* dst = unl_n + (size_t)b * CC * MU + m;
  for (int c = 0; c < CC; ++c) dst[(size_t)c * MU] = src[c * HWP] * inv;
}

// ---------------- Stage 1: u2s similarity GEMM + argmaxes (WMMA bf16) ----------------
__global__ void k_stage1(const float* __restrict__ unl_n, const float* __restrict__ sup_n,
                         int* __restrict__ u_near, float* __restrict__ cp1v, int* __restrict__ cp1i) {
  __shared__ float tile[8][256];
  __shared__ float cv[8][16];
  __shared__ int cr[8][16];
  int b = blockIdx.x, cy = blockIdx.y, tid = threadIdx.x;
  int wave = tid >> 5, lane = tid & 31, hi = lane >> 4, lo = lane & 15;
  int mbase = cy * 128 + wave * 16;
  int myrow = mbase + lo;
  bool rscan = (lane < 16);
  bool rvalid = rscan && (myrow < MU);
  float bestv = NEG_INF; int bestc = 0;

  int mA = imin(mbase + lo, MU - 1);  // clamped A row
  const float* apA = unl_n + (size_t)b * CC * MU + mA;

  for (int ct = 0; ct < 138; ++ct) {
    int colN = imin(ct * 16 + lo, NCOL1 - 1);  // clamped B col
    int n = colN / MS;
    int s = colN % MS;
    const float* bp = sup_n + ((size_t)(b * NWAY + n) * CC) * MS + s;
    v8f d = {0.f, 0.f, 0.f, 0.f, 0.f, 0.f, 0.f, 0.f};
    v16bf a0 = load_bf16_tile(apA, 0, MU, hi);
    v16bf b0 = load_bf16_tile(bp, 0, MS, hi);
    d = wmma_bf16(a0, b0, d);
    v16bf a1 = load_bf16_tile(apA, 32, MU, hi);
    v16bf b1 = load_bf16_tile(bp, 32, MS, hi);
    d = wmma_bf16(a1, b1, d);
#pragma unroll
    for (int i = 0; i < 8; ++i) tile[wave][(i + hi * 8) * 16 + lo] = d[i];
    __builtin_amdgcn_wave_barrier();
    if (rvalid) {
      for (int j = 0; j < 16; ++j) {
        int col = ct * 16 + j;
        if (col < NCOL1) {
          float v = tile[wave][lo * 16 + j];
          if (v > bestv) { bestv = v; bestc = col; }
        }
      }
    }
    float cmax = NEG_INF; int crw = 0x7fffffff;
#pragma unroll
    for (int i = 0; i < 8; ++i) {
      int row = mbase + i + hi * 8;
      if (row < MU && d[i] > cmax) { cmax = d[i]; crw = row; }
    }
    float ov = __shfl_xor(cmax, 16, 32);
    int orw = __shfl_xor(crw, 16, 32);
    if (ov > cmax || (ov == cmax && orw < crw)) { cmax = ov; crw = orw; }
    if (lane < 16) { cv[wave][lane] = cmax; cr[wave][lane] = crw; }
    __syncthreads();
    if (wave == 0 && lane < 16) {
      int col = ct * 16 + lane;
      if (col < NCOL1) {
        float mv = cv[0][lane]; int mr = cr[0][lane];
        for (int w = 1; w < 8; ++w) {
          float v = cv[w][lane];
          if (v > mv) { mv = v; mr = cr[w][lane]; }
        }
        size_t o = ((size_t)(b * CH1 + cy)) * NCOL1 + col;
        cp1v[o] = mv; cp1i[o] = mr;
      }
    }
    __syncthreads();
  }
  if (rvalid) u_near[b * MU + myrow] = bestc;
}

__global__ void k_colred1(const float* __restrict__ cp1v, const int* __restrict__ cp1i,
                          int* __restrict__ s_near1) {
  int idx = blockIdx.x * blockDim.x + threadIdx.x;
  if (idx >= BB * NCOL1) return;
  int b = idx / NCOL1, col = idx % NCOL1;
  float best = NEG_INF; int bi = 0;
  for (int ch = 0; ch < CH1; ++ch) {
    size_t o = ((size_t)(b * CH1 + ch)) * NCOL1 + col;
    float v = cp1v[o];
    if (v > best) { best = v; bi = cp1i[o]; }
  }
  s_near1[idx] = bi;
}

// Mutual-NN selection: ordered index lists per (b,n), counts, L, Lt.
__global__ void k_select(const int* __restrict__ u_near, const int* __restrict__ s_near1,
                         int* __restrict__ idxlist, int* __restrict__ counts, int* __restrict__ params) {
  int t = threadIdx.x;
  if (t < BB * NWAY) {
    int b = t / NWAY, n = t % NWAY;
    int cnt = 0;
    for (int m = 0; m < MU; ++m) {
      int un = u_near[b * MU + m];
      // cls == u_near/441 (first-index tie semantics), mutual: s_near[u_near]==m
      if (un / MS == n && s_near1[b * NCOL1 + un] == m) { idxlist[t * LCAP + cnt] = m; ++cnt; }
    }
    counts[t] = cnt;
  }
  __syncthreads();
  if (t == 0) {
    int L = 0;
    for (int i = 0; i < BB * NWAY; ++i) L = (counts[i] > L) ? counts[i] : L;
    params[0] = L;
    params[1] = MS + L;  // Lt
  }
}

// ---------------- Query projections + normalized value-query ----------------
__global__ void k_qproj(const float* __restrict__ qxf, const float* __restrict__ qw,
                        const float* __restrict__ vw, float* __restrict__ q_proj,
                        float* __restrict__ vq_n) {
  __shared__ float col[CC];
  __shared__ float red[CC];
  int m = blockIdx.x, bq = blockIdx.y, o = threadIdx.x;
  col[o] = qxf[((size_t)bq * CC + o) * HWP + m];
  __syncthreads();
  float qp = 0.f, vv = 0.f;
  for (int c = 0; c < CC; ++c) {
    float x = col[c];
    qp += qw[o * CC + c] * x;
    vv += vw[o * CC + c] * x;
  }
  red[o] = vv * vv;
  __syncthreads();
  if (o == 0) {
    float ss = 0.f;
    for (int i = 0; i < CC; ++i) ss += red[i];
    red[0] = sqrtf(ss) + 1e-16f;
  }
  __syncthreads();
  float inv = 1.0f / red[0];
  q_proj[((size_t)bq * CC + o) * HWP + m] = qp;
  vq_n[((size_t)bq * CC + o) * HWP + m] = vv * inv;
}

// ---------------- Support+unlabeled concat -> key/value projections ----------------
__global__ void k_supproj(const float* __restrict__ sxf, const float* __restrict__ uxf,
                          const float* __restrict__ kw, const float* __restrict__ vw,
                          const int* __restrict__ idxlist, const int* __restrict__ counts,
                          const int* __restrict__ params,
                          float* __restrict__ k_sup, float* __restrict__ v_sup) {
  __shared__ float col[CC];
  int l = blockIdx.x, bn = blockIdx.y, t = threadIdx.x;
  int Lt = params[1];
  if (l >= Lt) return;
  float x;
  if (l < MS) {
    x = sxf[((size_t)bn * CC + t) * HWP + l];
  } else {
    int i = l - MS;
    if (i < counts[bn]) {
      int m = idxlist[bn * LCAP + i];
      int b = bn / NWAY;
      int u = m / HWP, p = m % HWP;
      x = uxf[((size_t)(b * UU + u) * CC + t) * HWP + p];
    } else x = 0.f;  // zero padding column (participates in softmax/argmax)
  }
  col[t] = x;
  __syncthreads();
  float kacc = 0.f, vacc = 0.f;
  for (int c = 0; c < CC; ++c) {
    float y = col[c];
    kacc += kw[t * CC + c] * y;
    vacc += vw[t * CC + c] * y;
  }
  k_sup[((size_t)bn * CC + t) * LTMAX + l] = kacc;
  v_sup[((size_t)bn * CC + t) * LTMAX + l] = vacc;
}

__global__ void k_vmean(const float* __restrict__ v_sup, const int* __restrict__ params,
                        float* __restrict__ v_mean) {
  int idx = blockIdx.x * blockDim.x + threadIdx.x;
  if (idx >= BB * NWAY * CC) return;
  int Lt = params[1];
  const float* p = v_sup + (size_t)idx * LTMAX;
  float s = 0.f;
  for (int l = 0; l < Lt; ++l) s += p[l];
  v_mean[idx] = s / (float)Lt;
}

// ---------------- Fused attention: WMMA scores + online softmax + P@V + argmax stats ----------------
__global__ void k_attn(const float* __restrict__ q_proj, const float* __restrict__ k_sup,
                       const float* __restrict__ v_sup, const int* __restrict__ params,
                       float* __restrict__ aligned, int* __restrict__ q_near,
                       float* __restrict__ cp2v, int* __restrict__ cp2i) {
  __shared__ float tile[8][256];      // per-wave score/P tile
  __shared__ float Vt[16 * CC];       // shared V tile (16 l x 64 o)
  __shared__ float acc[8][16 * CC];   // per-wave output accumulators
  __shared__ float scaleR[8][16];
  __shared__ float rowDsh[8][16];
  __shared__ float cv[8][16];
  __shared__ int cr[8][16];

  int bq = blockIdx.x, cy = blockIdx.y;
  int b = bq / QQ;
  int tid = threadIdx.x;
  int wave = tid >> 5, lane = tid & 31, hi = lane >> 4, lo = lane & 15;
  int mbase = cy * 128 + wave * 16;
  int Lt = params[1];
  int NT = (Lt + 15) >> 4;

  int mA = imin(mbase + lo, HWP - 1);  // clamped A row
  const float* apA = q_proj + (size_t)bq * CC * HWP + mA;

  bool rscan = lane < 16;
  int myrow = mbase + lo;
  bool rvalid = rscan && myrow < HWP;
  float qbestv = NEG_INF; int qbestc = 0;

  for (int n = 0; n < NWAY; ++n) {
    int bn = b * NWAY + n;
    const float* kp = k_sup + (size_t)bn * CC * LTMAX;
    const float* vp = v_sup + (size_t)bn * CC * LTMAX;
    float rowM = NEG_INF, rowD = 0.f;
    for (int t2 = 0; t2 < 32; ++t2) acc[wave][lane * 32 + t2] = 0.f;

    for (int lt = 0; lt < NT; ++lt) {
      int lbase = lt * 16;
      __syncthreads();
      for (int r = 0; r < 4; ++r) {  // cooperative V tile load (clamped; P=0 kills pad cols)
        int s = tid + r * 256;
        int lv = s >> 6, o = s & 63;
        int l = imin(lbase + lv, Lt - 1);
        Vt[s] = vp[(size_t)o * LTMAX + l];
      }
      __syncthreads();
      int lB = imin(lbase + lo, Lt - 1);  // clamped B col
      const float* bp = kp + lB;
      v8f d = {0.f, 0.f, 0.f, 0.f, 0.f, 0.f, 0.f, 0.f};
      v16bf a0 = load_bf16_tile(apA, 0, HWP, hi);
      v16bf bb0 = load_bf16_tile(bp, 0, LTMAX, hi);
      d = wmma_bf16(a0, bb0, d);
      v16bf a1 = load_bf16_tile(apA, 32, HWP, hi);
      v16bf bb1 = load_bf16_tile(bp, 32, LTMAX, hi);
      d = wmma_bf16(a1, bb1, d);
#pragma unroll
      for (int i = 0; i < 8; ++i) d[i] *= 0.125f;  // / sqrt(PROJ_DIM)
#pragma unroll
      for (int i = 0; i < 8; ++i) tile[wave][(i + hi * 8) * 16 + lo] = d[i];
      __builtin_amdgcn_wave_barrier();
      if (rscan) {  // per-row: argmax over merged cols + online softmax, P in-place
        float tmax = NEG_INF;
        for (int j = 0; j < 16; ++j) {
          int l = lbase + j;
          if (l < Lt) {
            float v = tile[wave][lo * 16 + j];
            if (rvalid && v > qbestv) { qbestv = v; qbestc = n * LTMAX + l; }
            if (v > tmax) tmax = v;
          }
        }
        float newM = fmaxf(rowM, tmax);
        float scale = __expf(rowM - newM);
        float se = 0.f;
        for (int j = 0; j < 16; ++j) {
          int l = lbase + j;
          float e = 0.f;
          if (l < Lt) e = __expf(tile[wave][lo * 16 + j] - newM);
          tile[wave][lo * 16 + j] = e;
          se += e;
        }
        rowD = rowD * scale + se;
        rowM = newM;
        scaleR[wave][lo] = scale;
      }
      __builtin_amdgcn_wave_barrier();
      float cmax = NEG_INF; int crw = 0x7fffffff;  // column-side from registers
#pragma unroll
      for (int i = 0; i < 8; ++i) {
        int row = mbase + i + hi * 8;
        if (row < HWP && d[i] > cmax) { cmax = d[i]; crw = row; }
      }
      float ovv = __shfl_xor(cmax, 16, 32);
      int orw = __shfl_xor(crw, 16, 32);
      if (ovv > cmax || (ovv == cmax && orw < crw)) { cmax = ovv; crw = orw; }
      if (lane < 16) { cv[wave][lane] = cmax; cr[wave][lane] = crw; }
      __syncthreads();
      for (int t2 = 0; t2 < 32; ++t2) {  // acc = acc*scale + P@V
        int slot = lane * 32 + t2;
        int r = slot >> 6, o = slot & 63;
        float s = 0.f;
        for (int j = 0; j < 16; ++j) s += tile[wave][r * 16 + j] * Vt[j * 64 + o];
        acc[wave][slot] = acc[wave][slot] * scaleR[wave][r] + s;
      }
      if (wave == 0 && lane < 16) {  // block col-argmax partial
        int l = lbase + lane;
        if (l < Lt) {
          float mv = cv[0][lane]; int mr = cr[0][lane];
          for (int w = 1; w < 8; ++w) {
            float v = cv[w][lane];
            if (v > mv) { mv = v; mr = cr[w][lane]; }
          }
          size_t oo = (((size_t)(bq * CH2 + cy)) * NWAY + n) * LTMAX + l;
          cp2v[oo] = mv; cp2i[oo] = mr;
        }
      }
    }  // lt
    if (rscan) rowDsh[wave][lo] = rowD;
    __builtin_amdgcn_wave_barrier();
    for (int t2 = 0; t2 < 32; ++t2) {
      int slot = lane * 32 + t2;
      int r = slot >> 6, o = slot & 63;
      int row = mbase + r;
      if (row < HWP)
        aligned[(((size_t)bq * NWAY + n) * HWP + row) * CC + o] = acc[wave][slot] / rowDsh[wave][r];
    }
  }  // n
  if (rvalid) q_near[bq * HWP + myrow] = qbestc;
}

__global__ void k_colred2(const float* __restrict__ cp2v, const int* __restrict__ cp2i,
                          const int* __restrict__ params, int* __restrict__ s_near2) {
  int idx = blockIdx.x * blockDim.x + threadIdx.x;
  if (idx >= BB * QQ * NWAY * LTMAX) return;
  int Lt = params[1];
  int l = idx % LTMAX;
  if (l >= Lt) return;
  int rest = idx / LTMAX;  // bq*NWAY + n
  int n = rest % NWAY; int bq = rest / NWAY;
  float best = NEG_INF; int bi = 0;
  for (int ch = 0; ch < CH2; ++ch) {
    size_t o = (((size_t)(bq * CH2 + ch)) * NWAY + n) * LTMAX + l;
    float v = cp2v[o];
    if (v > best) { best = v; bi = cp2i[o]; }
  }
  s_near2[(size_t)rest * LTMAX + l] = bi;
}

// q_mask select (raw softmax vs uniform mean) + L2 normalize, in place.
__global__ void k_finalize_aligned(const int* __restrict__ q_near, const int* __restrict__ s_near2,
                                   const float* __restrict__ v_mean, float* __restrict__ aligned) {
  int idx = blockIdx.x * blockDim.x + threadIdx.x;
  if (idx >= BB * QQ * NWAY * HWP) return;
  int m = idx % HWP;
  int rest = idx / HWP;  // bq*NWAY + n
  int n = rest % NWAY; int bq = rest / NWAY;
  int b = bq / QQ;
  int col = q_near[bq * HWP + m];
  bool mask = (s_near2[(size_t)bq * NWAY * LTMAX + col] == m);
  float* p = aligned + (size_t)idx * CC;
  const float* vm = v_mean + (size_t)(b * NWAY + n) * CC;
  float ss = 0.f;
  for (int o = 0; o < CC; ++o) { float x = mask ? p[o] : vm[o]; ss += x * x; }
  float inv = 1.0f / (sqrtf(ss) + 1e-16f);
  for (int o = 0; o < CC; ++o) { float x = mask ? p[o] : vm[o]; p[o] = x * inv; }
}

// ---------------- Logits: aligned @ vq^T with column-max + sum (WMMA bf16) ----------------
__global__ void k_logits(const float* __restrict__ aligned, const float* __restrict__ vq_n,
                         float* __restrict__ logits) {
  __shared__ float red[8][16];
  __shared__ float sums[16];
  int bqn = blockIdx.x;
  int bq = bqn / NWAY;
  int tid = threadIdx.x;
  int wave = tid >> 5, lane = tid & 31, hi = lane >> 4, lo = lane & 15;
  float partial = 0.f;
  for (int ct = 0; ct < 28; ++ct) {
    int k = imin(ct * 16 + lo, HWP - 1);  // clamped B col
    const float* bp = vq_n + (size_t)bq * CC * HWP + k;
    float cmax = NEG_INF;
    for (int mt = wave; mt < 28; mt += 8) {
      int m = imin(mt * 16 + lo, HWP - 1);  // clamped A row
      const float* ap = aligned + ((size_t)bqn * HWP + m) * CC;
      v8f d = {0.f, 0.f, 0.f, 0.f, 0.f, 0.f, 0.f, 0.f};
      v16bf a0 = load_bf16_tile(ap, 0, 1, hi);
      v16bf b0 = load_bf16_tile(bp, 0, HWP, hi);
      d = wmma_bf16(a0, b0, d);
      v16bf a1 = load_bf16_tile(ap, 32, 1, hi);
      v16bf b1 = load_bf16_tile(bp, 32, HWP, hi);
      d = wmma_bf16(a1, b1, d);
#pragma unroll
      for (int i = 0; i < 8; ++i) {
        int row = mt * 16 + i + hi * 8;
        if (row < HWP) cmax = fmaxf(cmax, d[i]);
      }
    }
    float o2 = __shfl_xor(cmax, 16, 32);
    cmax = fmaxf(cmax, o2);
    if (lane < 16) red[wave][lane] = cmax;
    __syncthreads();
    if (tid < 16) {
      float mm = red[0][tid];
      for (int w = 1; w < 8; ++w) mm = fmaxf(mm, red[w][tid]);
      int kk = ct * 16 + tid;
      if (kk < HWP) partial += mm;
    }
    __syncthreads();
  }
  if (tid < 16) sums[tid] = partial;
  __syncthreads();
  if (tid == 0) {
    float t = 0.f;
    for (int i = 0; i < 16; ++i) t += sums[i];
    logits[bqn] = 0.5f * ((float)HWP + t);  // sum_k 0.5*(1+max_m dot)
  }
}

__global__ void k_loss(const float* __restrict__ logits, const int* __restrict__ qy,
                       float* __restrict__ out) {
  if (threadIdx.x != 0 || blockIdx.x != 0) return;
  float acc = 0.f;
  for (int i = 0; i < BB * QQ; ++i) {
    float z[NWAY]; float mx = NEG_INF;
    for (int j = 0; j < NWAY; ++j) { z[j] = logits[i * NWAY + j] * 5.0f; mx = fmaxf(mx, z[j]); }
    float s = 0.f;
    for (int j = 0; j < NWAY; ++j) s += __expf(z[j] - mx);
    int y = qy[i];
    acc -= (z[y] - mx - logf(s));
  }
  out[0] = acc / (float)(BB * QQ);
}

extern "C" void kernel_launch(void* const* d_in, const int* in_sizes, int n_in,
                              void* d_out, int out_size, void* d_ws, size_t ws_size,
                              hipStream_t stream) {
  (void)in_sizes; (void)n_in; (void)out_size; (void)ws_size;
  const float* sxf = (const float*)d_in[0];
  const float* qxf = (const float*)d_in[2];
  const int* qy = (const int*)d_in[3];
  const float* uxf = (const float*)d_in[4];
  const float* kw = (const float*)d_in[5];
  const float* qw = (const float*)d_in[6];
  const float* vw = (const float*)d_in[7];

  char* base = (char*)d_ws; size_t off = 0;
  auto carve = [&](size_t bytes) -> void* {
    void* p = base + off;
    off += (bytes + 255) & ~(size_t)255;
    return p;
  };
  float* sup_n  = (float*)carve(sizeof(float) * BB * NWAY * CC * MS);
  float* unl_n  = (float*)carve(sizeof(float) * (size_t)BB * CC * MU);
  float* cp1v   = (float*)carve(sizeof(float) * (size_t)BB * CH1 * NCOL1);
  int*   cp1i   = (int*)carve(sizeof(int) * (size_t)BB * CH1 * NCOL1);
  int*   u_near = (int*)carve(sizeof(int) * BB * MU);
  int*   s_near1= (int*)carve(sizeof(int) * BB * NCOL1);
  int*   counts = (int*)carve(sizeof(int) * BB * NWAY);
  int*   params = (int*)carve(sizeof(int) * 4);
  int*   idxlist= (int*)carve(sizeof(int) * BB * NWAY * LCAP);
  float* ksup   = (float*)carve(sizeof(float) * (size_t)BB * NWAY * CC * LTMAX);
  float* vsup   = (float*)carve(sizeof(float) * (size_t)BB * NWAY * CC * LTMAX);
  float* vmean  = (float*)carve(sizeof(float) * BB * NWAY * CC);
  float* qproj  = (float*)carve(sizeof(float) * (size_t)BB * QQ * CC * HWP);
  float* vqn    = (float*)carve(sizeof(float) * (size_t)BB * QQ * CC * HWP);
  float* cp2v   = (float*)carve(sizeof(float) * (size_t)BB * QQ * CH2 * NWAY * LTMAX);
  int*   cp2i   = (int*)carve(sizeof(int) * (size_t)BB * QQ * CH2 * NWAY * LTMAX);
  int*   qnear  = (int*)carve(sizeof(int) * BB * QQ * HWP);
  int*   snear2 = (int*)carve(sizeof(int) * (size_t)BB * QQ * NWAY * LTMAX);
  float* aligned= (float*)carve(sizeof(float) * (size_t)BB * QQ * NWAY * HWP * CC);
  float* logits = (float*)carve(sizeof(float) * BB * QQ * NWAY);

  k_norm_sup<<<(BB * NWAY * MS + 255) / 256, 256, 0, stream>>>(sxf, sup_n);
  k_norm_unl<<<(BB * MU + 255) / 256, 256, 0, stream>>>(uxf, unl_n);
  k_stage1<<<dim3(BB, CH1), 256, 0, stream>>>(unl_n, sup_n, u_near, cp1v, cp1i);
  k_colred1<<<(BB * NCOL1 + 255) / 256, 256, 0, stream>>>(cp1v, cp1i, s_near1);
  k_select<<<1, 32, 0, stream>>>(u_near, s_near1, idxlist, counts, params);
  k_qproj<<<dim3(HWP, BB * QQ), CC, 0, stream>>>(qxf, qw, vw, qproj, vqn);
  k_supproj<<<dim3(LTMAX, BB * NWAY), CC, 0, stream>>>(sxf, uxf, kw, vw, idxlist, counts, params, ksup, vsup);
  k_vmean<<<(BB * NWAY * CC + 63) / 64, 64, 0, stream>>>(vsup, params, vmean);
  k_attn<<<dim3(BB * QQ, CH2), 256, 0, stream>>>(qproj, ksup, vsup, params, aligned, qnear, cp2v, cp2i);
  k_colred2<<<(BB * QQ * NWAY * LTMAX + 255) / 256, 256, 0, stream>>>(cp2v, cp2i, params, snear2);
  k_finalize_aligned<<<(BB * QQ * NWAY * HWP + 255) / 256, 256, 0, stream>>>(qnear, snear2, vmean, aligned);
  k_logits<<<BB * QQ * NWAY, 256, 0, stream>>>(aligned, vqn, logits);
  k_loss<<<1, 32, 0, stream>>>(logits, qy, (float*)d_out);
}
